// SparseExpert_64123861729522
// MI455X (gfx1250) — matrix operand order repaired and checked by reading, not compiled, not run on hardware
//
#include <hip/hip_runtime.h>
#include <stdint.h>

// Problem constants (match reference setup_inputs)
#define T_TOK 8192   // B*S
#define DM    1024   // D
#define NE    8      // experts
#define TOPK  2

typedef __attribute__((ext_vector_type(8)))  __bf16 v8bf;
typedef __attribute__((ext_vector_type(16))) __bf16 v16bf;
typedef __attribute__((ext_vector_type(8)))  float  v8f;
typedef __attribute__((__vector_size__(16))) int    vsi4;   // 4 x i32, matches builtin

#if __has_builtin(__builtin_amdgcn_global_load_async_to_lds_b128)
#define HAVE_ASYNC 1
#else
#define HAVE_ASYNC 0
#endif

__device__ __forceinline__ __bf16 f2bf(float f) {
  unsigned u = __float_as_uint(f);
  unsigned r = (u + 0x7fffu + ((u >> 16) & 1u)) >> 16;   // RNE
  unsigned short s = (unsigned short)r;
  return *reinterpret_cast<__bf16*>(&s);
}

// ---------------------------------------------------------------------------
// 1) f32 -> bf16 conversion (x and W), 4 elements/thread
// ---------------------------------------------------------------------------
__global__ __launch_bounds__(256) void moe_cvt_bf16(const float* __restrict__ src,
                                                    __bf16* __restrict__ dst, int n4) {
  int i = blockIdx.x * 256 + threadIdx.x;
  if (i >= n4) return;
  float4 v = reinterpret_cast<const float4*>(src)[i];
  union { __bf16 h[4]; unsigned long long u; } pk;
  pk.h[0] = f2bf(v.x); pk.h[1] = f2bf(v.y); pk.h[2] = f2bf(v.z); pk.h[3] = f2bf(v.w);
  reinterpret_cast<unsigned long long*>(dst)[i] = pk.u;
}

// ---------------------------------------------------------------------------
// 2) Router: one wave32 per token. Writes logits, top-2 (sel, renorm weight),
//    expert histogram.
// ---------------------------------------------------------------------------
__global__ __launch_bounds__(256) void moe_router(const float* __restrict__ x,
                                                  const float* __restrict__ gw,
                                                  const float* __restrict__ gb,
                                                  float* __restrict__ logits_out,
                                                  int* __restrict__ sel,
                                                  float* __restrict__ selw,
                                                  int* __restrict__ counts) {
  __shared__ float sgw[NE * DM];
  int tid = threadIdx.x;
  for (int i = tid; i < NE * DM; i += 256) sgw[i] = gw[i];
  __syncthreads();

  int wave = tid >> 5, lane = tid & 31;
  int t = blockIdx.x * 8 + wave;
  const float* xr = x + (size_t)t * DM;

  float acc[NE];
#pragma unroll
  for (int e = 0; e < NE; ++e) acc[e] = 0.f;

  for (int d = lane; d < DM; d += 32) {
    float xv = xr[d];
#pragma unroll
    for (int e = 0; e < NE; ++e) acc[e] += xv * sgw[e * DM + d];
  }
#pragma unroll
  for (int e = 0; e < NE; ++e) {
#pragma unroll
    for (int off = 16; off > 0; off >>= 1) acc[e] += __shfl_xor(acc[e], off, 32);
  }

  if (lane == 0) {
    float l[NE];
#pragma unroll
    for (int e = 0; e < NE; ++e) l[e] = acc[e] + gb[e];
#pragma unroll
    for (int e = 0; e < NE; ++e) logits_out[(size_t)t * NE + e] = l[e];
    // top-2 (lowest index wins ties, like jax top_k)
    int i0 = 0;
#pragma unroll
    for (int e = 1; e < NE; ++e) if (l[e] > l[i0]) i0 = e;
    int i1 = (i0 == 0) ? 1 : 0;
#pragma unroll
    for (int e = 0; e < NE; ++e) if (e != i0 && l[e] > l[i1]) i1 = e;
    // softmax denom cancels in renorm: w0 = 1/(1+e1), w1 = e1/(1+e1)
    float e1 = __expf(l[i1] - l[i0]);
    float inv = 1.f / (1.f + e1);
    sel[2 * t + 0] = i0;  selw[2 * t + 0] = inv;
    sel[2 * t + 1] = i1;  selw[2 * t + 1] = e1 * inv;
    atomicAdd(&counts[i0], 1);
    atomicAdd(&counts[i1], 1);
  }
}

// ---------------------------------------------------------------------------
// 3) Exclusive scan over 8 expert counts
// ---------------------------------------------------------------------------
__global__ void moe_scan8(const int* __restrict__ counts, int* __restrict__ offs,
                          int* __restrict__ curs) {
  if (threadIdx.x == 0 && blockIdx.x == 0) {
    int s = 0;
    for (int e = 0; e < NE; ++e) { offs[e] = s; curs[e] = s; s += counts[e]; }
  }
}

// ---------------------------------------------------------------------------
// 4) Scatter token ids + routing weights into per-expert buckets
// ---------------------------------------------------------------------------
__global__ __launch_bounds__(256) void moe_scatter(const int* __restrict__ sel,
                                                   const float* __restrict__ selw,
                                                   int* __restrict__ curs,
                                                   int* __restrict__ tok,
                                                   float* __restrict__ twgt) {
  int t = blockIdx.x * 256 + threadIdx.x;
  if (t >= T_TOK) return;
#pragma unroll
  for (int k = 0; k < TOPK; ++k) {
    int e = sel[2 * t + k];
    int p = atomicAdd(&curs[e], 1);
    tok[p] = t;
    twgt[p] = selw[2 * t + k];
  }
}

// ---------------------------------------------------------------------------
// 5) Grouped per-expert GEMM with bf16 WMMA, double-buffered async LDS staging.
//    Block tile: 128(M) x 64(N), K step 32. 8 waves as 4x2, each 32x32.
// ---------------------------------------------------------------------------
#define BM 128
#define BN 64
#define BK 32
#define LPAD 8
#define LSTR (BK + LPAD)   // 40 bf16 -> 80 B row stride (16B aligned)

__device__ __forceinline__ void copy16_g2s(const __bf16* g, __bf16* s) {
#if HAVE_ASYNC
  __builtin_amdgcn_global_load_async_to_lds_b128(
      (__attribute__((address_space(1))) vsi4*)g,
      (__attribute__((address_space(3))) vsi4*)s, 0, 0);
#else
  *reinterpret_cast<v8bf*>(s) = *reinterpret_cast<const v8bf*>(g);
#endif
}

template <int N>
__device__ __forceinline__ void async_wait() {
#if HAVE_ASYNC
#if __has_builtin(__builtin_amdgcn_s_wait_asynccnt)
  __builtin_amdgcn_s_wait_asynccnt(N);
#else
  asm volatile("s_wait_asynccnt %0" :: "i"(N) : "memory");
#endif
#endif
}

__global__ __launch_bounds__(256) void moe_gemm(const __bf16* __restrict__ xb,
                                                const __bf16* __restrict__ Wb,
                                                const float* __restrict__ bias,
                                                const int* __restrict__ tok,
                                                const float* __restrict__ twgt,
                                                const int* __restrict__ offs,
                                                const int* __restrict__ cnts,
                                                float* __restrict__ out) {
  int e = blockIdx.z;
  int cnt = cnts[e];
  int m0 = blockIdx.y * BM;
  if (m0 >= cnt) return;                 // expert bucket exhausted
  int n0 = blockIdx.x * BN;
  int base = offs[e];

  __shared__ __bf16 As[2][BM][LSTR];
  __shared__ __bf16 Bs[2][BN][LSTR];
  __shared__ int    stok[BM];
  __shared__ float  swgt[BM];

  int tid = threadIdx.x;
  if (tid < BM) {
    int m = m0 + tid;
    bool valid = m < cnt;
    stok[tid] = valid ? tok[base + m] : 0;
    swgt[tid] = valid ? twgt[base + m] : 0.f;
  }
  __syncthreads();

  int wave = tid >> 5, lane = tid & 31;
  int wm = wave >> 1;        // 0..3 -> M offset wm*32
  int wn = wave & 1;         // 0..1 -> N offset wn*32

  v8f acc[2][2];
  v8f zero = {0.f, 0.f, 0.f, 0.f, 0.f, 0.f, 0.f, 0.f};
#pragma unroll
  for (int i = 0; i < 2; ++i)
#pragma unroll
    for (int j = 0; j < 2; ++j) acc[i][j] = zero;

  int lrow = tid >> 2;               // 0..63
  int lcol = (tid & 3) * 8;          // bf16 offset in [0,32)
  const int mr  = lane & 15;
  const int akb = (lane < 16) ? 0 : 8;    // A: K base per half-wave
  const int bkb = (lane < 16) ? 0 : 8;    // B: K base per half-wave

  // Hoist gather bases out of the K loop (token rows are loop-invariant).
  const __bf16* ga0 = xb + (size_t)stok[lrow] * DM + lcol;
  const __bf16* ga1 = xb + (size_t)stok[lrow + 64] * DM + lcol;
  const __bf16* gwb = Wb + ((size_t)e * DM + n0 + lrow) * DM + lcol;

  constexpr int NK = DM / BK;        // 32 K-steps

  // Prologue: stage tile 0 into buffer 0.
  copy16_g2s(ga0, &As[0][lrow][lcol]);
  copy16_g2s(ga1, &As[0][lrow + 64][lcol]);
  copy16_g2s(gwb, &Bs[0][lrow][lcol]);

#pragma unroll 2
  for (int kt = 0; kt < NK; ++kt) {
    int cur = kt & 1;
    if (kt + 1 < NK) {
      int nxt = (kt + 1) & 1;
      int k = (kt + 1) * BK;
      copy16_g2s(ga0 + k, &As[nxt][lrow][lcol]);
      copy16_g2s(ga1 + k, &As[nxt][lrow + 64][lcol]);
      copy16_g2s(gwb + k, &Bs[nxt][lrow][lcol]);
      async_wait<3>();                 // current group done; prefetch in flight
    } else {
      async_wait<0>();
    }
    __syncthreads();

    union { v16bf v; v8bf h[2]; } a0, a1, b0, b1;
    a0.h[0] = *reinterpret_cast<const v8bf*>(&As[cur][wm * 32 + mr][akb]);
    a0.h[1] = *reinterpret_cast<const v8bf*>(&As[cur][wm * 32 + mr][akb + 16]);
    a1.h[0] = *reinterpret_cast<const v8bf*>(&As[cur][wm * 32 + 16 + mr][akb]);
    a1.h[1] = *reinterpret_cast<const v8bf*>(&As[cur][wm * 32 + 16 + mr][akb + 16]);
    b0.h[0] = *reinterpret_cast<const v8bf*>(&Bs[cur][wn * 32 + mr][bkb]);
    b0.h[1] = *reinterpret_cast<const v8bf*>(&Bs[cur][wn * 32 + mr][bkb + 16]);
    b1.h[0] = *reinterpret_cast<const v8bf*>(&Bs[cur][wn * 32 + 16 + mr][bkb]);
    b1.h[1] = *reinterpret_cast<const v8bf*>(&Bs[cur][wn * 32 + 16 + mr][bkb + 16]);

    acc[0][0] = __builtin_amdgcn_wmma_f32_16x16x32_bf16(false, a0.v, false, b0.v,
                                                        (short)0, acc[0][0], false, false);
    acc[0][1] = __builtin_amdgcn_wmma_f32_16x16x32_bf16(false, a0.v, false, b1.v,
                                                        (short)0, acc[0][1], false, false);
    acc[1][0] = __builtin_amdgcn_wmma_f32_16x16x32_bf16(false, a1.v, false, b0.v,
                                                        (short)0, acc[1][0], false, false);
    acc[1][1] = __builtin_amdgcn_wmma_f32_16x16x32_bf16(false, a1.v, false, b1.v,
                                                        (short)0, acc[1][1], false, false);
    __syncthreads();   // protect buffer `cur` from the issue in iteration kt+1
  }

  // Epilogue: out[t] += w * (acc + bias[e])   (token appears under 2 experts)
#pragma unroll
  for (int im = 0; im < 2; ++im) {
#pragma unroll
    for (int in = 0; in < 2; ++in) {
      int ng = n0 + wn * 32 + in * 16 + (lane & 15);
      float bv = bias[e * DM + ng];
#pragma unroll
      for (int r = 0; r < 8; ++r) {
        int ml = wm * 32 + im * 16 + r + ((lane >= 16) ? 8 : 0);
        if (m0 + ml < cnt) {
          int t = stok[ml];
          atomicAdd(out + (size_t)t * DM + ng, swgt[ml] * (acc[im][in][r] + bv));
        }
      }
    }
  }
}

// ---------------------------------------------------------------------------
extern "C" void kernel_launch(void* const* d_in, const int* in_sizes, int n_in,
                              void* d_out, int out_size, void* d_ws, size_t ws_size,
                              hipStream_t stream) {
  const float* x  = (const float*)d_in[0];   // [T, D]
  const float* gw = (const float*)d_in[1];   // [E, D]
  const float* gb = (const float*)d_in[2];   // [E]
  const float* W  = (const float*)d_in[3];   // [E, D, D]
  const float* b  = (const float*)d_in[4];   // [E, D]

  float* out    = (float*)d_out;                       // [T, D]
  float* logits = out + (size_t)T_TOK * DM;            // [T, E]

  // Workspace layout
  char* p = (char*)d_ws;
  __bf16* xb = (__bf16*)p;                 p += (size_t)T_TOK * DM * 2;   // 16 MB
  __bf16* Wb = (__bf16*)p;                 p += (size_t)NE * DM * DM * 2; // 16 MB
  int*   sel    = (int*)p;                 p += (size_t)T_TOK * 2 * 4;
  float* selw   = (float*)p;               p += (size_t)T_TOK * 2 * 4;
  int*   tok    = (int*)p;                 p += (size_t)T_TOK * 2 * 4;
  float* twgt   = (float*)p;               p += (size_t)T_TOK * 2 * 4;
  int*   counts = (int*)p;                 p += 8 * 4;
  int*   offs   = (int*)p;                 p += 8 * 4;
  int*   curs   = (int*)p;                 p += 8 * 4;

  // Accumulated via atomics -> zero output region; zero expert histogram.
  (void)hipMemsetAsync(out, 0, (size_t)T_TOK * DM * sizeof(float), stream);
  (void)hipMemsetAsync(counts, 0, 8 * sizeof(int), stream);

  int n4 = T_TOK * DM / 4;                   // == NE*DM*DM/4 too
  moe_cvt_bf16<<<(n4 + 255) / 256, 256, 0, stream>>>(x, xb, n4);
  moe_cvt_bf16<<<(n4 + 255) / 256, 256, 0, stream>>>(W, Wb, n4);

  moe_router<<<T_TOK / 8, 256, 0, stream>>>(x, gw, gb, logits, sel, selw, counts);
  moe_scan8<<<1, 32, 0, stream>>>(counts, offs, curs);
  moe_scatter<<<(T_TOK + 255) / 256, 256, 0, stream>>>(sel, selw, curs, tok, twgt);

  dim3 grid(DM / BN, T_TOK / BM, NE);        // (16, 64, 8); blocks past count exit early
  moe_gemm<<<grid, 256, 0, stream>>>(xb, Wb, b, tok, twgt, offs, counts, out);
}
